// AlignmentModel_91233695302504
// MI455X (gfx1250) — matrix operand order repaired and checked by reading, not compiled, not run on hardware
//
#include <hip/hip_runtime.h>
#include <hip/hip_bf16.h>

typedef __attribute__((ext_vector_type(16))) _Float16 v16h;
typedef __attribute__((ext_vector_type(8)))  float    v8f;
typedef __attribute__((ext_vector_type(4)))  unsigned int u32x4;
typedef __attribute__((ext_vector_type(8)))  int i32x8;
typedef __attribute__((ext_vector_type(4)))  int i32x4;

#define BS 16
#define TT 1024
#define DF 80
#define SZ 4096

#if __has_builtin(__builtin_amdgcn_tensor_load_to_lds) && __has_builtin(__builtin_amdgcn_s_wait_tensorcnt)
#define USE_TDM 1
#else
#define USE_TDM 0
#endif

// ---------------------------------------------------------------------------
// Kernel 1: centers c[b,t] = d/2 + roll(cumsum(d),1) with row-0 zero quirk.
// One wave per batch row; Kogge-Stone scan across 32 lanes x 32 elems/lane.
// ---------------------------------------------------------------------------
__global__ __launch_bounds__(32) void centers_kernel(const float* __restrict__ d,
                                                     float* __restrict__ c) {
  const int b    = blockIdx.x;
  const int lane = threadIdx.x;            // 0..31 (wave32)
  const float* db = d + (size_t)b * TT;

  float vals[32];
  float tot = 0.f;
#pragma unroll
  for (int j = 0; j < 32; ++j) { vals[j] = db[lane * 32 + j]; tot += vals[j]; }

  float incl = tot;
#pragma unroll
  for (int off = 1; off < 32; off <<= 1) {
    float v = __shfl_up(incl, off, 32);
    if (lane >= off) incl += v;
  }
  const float excl  = incl - tot;
  const float total = __shfl(incl, 31, 32);

  float run = excl;
#pragma unroll
  for (int j = 0; j < 32; ++j) {
    const int t = lane * 32 + j;
    // roll(cumsum,1): rolled[t] = excl-prefix for t>=1, rolled[0] = total sum.
    // Batch row 0 forced to zero (torch-faithful quirk).
    float base = (b == 0) ? 0.f : ((t == 0) ? total : run);
    c[(size_t)b * TT + t] = vals[j] * 0.5f + base;
    run += vals[j];
  }
}

// ---------------------------------------------------------------------------
// Kernel 2: hT[b][n][t] = (f16) h[b][t][n]  (transpose + downconvert once)
// ---------------------------------------------------------------------------
__global__ __launch_bounds__(256) void hT_kernel(const float* __restrict__ h,
                                                 _Float16* __restrict__ hT) {
  int idx = blockIdx.x * blockDim.x + threadIdx.x;
  const int tot = BS * DF * TT;
  if (idx >= tot) return;
  const int t = idx & (TT - 1);
  const int r = idx >> 10;                 // b*DF + n
  const int n = r % DF;
  const int b = r / DF;
  hT[idx] = (_Float16)h[((size_t)b * TT + t) * DF + n];
}

// ---------------------------------------------------------------------------
// LDS layout (single __shared__ struct -> group segment offset 0)
// ---------------------------------------------------------------------------
struct alignas(16) Smem {
  float    c[TT];            // offset 0,    4096 B
  _Float16 hb[2][DF * 32];   // offset 4096, 2 x 5120 B  ([n][32] halves per buf)
};
#define SMEM_C_OFF  0u
#define SMEM_HB_OFF 4096u
#define SMEM_HB_SZ  (DF * 32 * 2)   // 5120 bytes per buffer

#if USE_TDM
// 2D TDM descriptor: tensor = hT[b] (dim0 = t: len 1024 stride 1024, dim1 = n: len 80),
// tile = 32 x 80, element 2B.  DMA packs tile rows contiguously into LDS -> [n][32].
__device__ __forceinline__ void tdm_load_tile(const _Float16* gsrc, unsigned lds_off) {
  unsigned long long ga = (unsigned long long)(uintptr_t)gsrc;
  u32x4 g0;
  g0.x = 1u;                                              // count=1 (valid), user mode
  g0.y = lds_off;                                         // lds_addr (bytes)
  g0.z = (unsigned)(ga & 0xffffffffu);                    // global_addr[31:0]
  g0.w = (unsigned)((ga >> 32) & 0x01ffffffu) | (2u << 30); // addr[56:32] | type=2
  i32x8 g1;
  g1[0] = (1 << 16);          // workgroup_mask=0 | data_size=1 (2 bytes)
  g1[1] = (int)(TT << 16);    // tensor_dim0 = 1024  (bits 63:48)
  g1[2] = (DF << 16);         // tensor_dim1 = 80    (bits 95:80)
  g1[3] = (32 << 16);         // tile_dim0   = 32    (bits 127:112)
  g1[4] = DF;                 // tile_dim1   = 80    (bits 143:128), tile_dim2=0
  g1[5] = TT;                 // tensor_dim0_stride = 1024 (bits 191:160)
  g1[6] = 0;                  // stride hi / dim1_stride lo
  g1[7] = 0;
  i32x4 z4 = {0, 0, 0, 0};            // groups 2/3 unused for 2D tiles
  i32x8 z8 = {0, 0, 0, 0, 0, 0, 0, 0};
  // clang-23 / therock signature: (g0, g1, g2, g3, g4, cpol)
  __builtin_amdgcn_tensor_load_to_lds(g0, g1, z4, z4, z8, 0);
}
#endif

// ---------------------------------------------------------------------------
// Kernel 3: per-wave 16(s) x 80(feat) tile; W built in registers in WMMA
// A-layout; B staged once per block via TDM (double-buffered), 10 WMMA / step.
// ---------------------------------------------------------------------------
__global__ __launch_bounds__(128) void align_kernel(const _Float16* __restrict__ hT,
                                                    const float* __restrict__ c,
                                                    const float* __restrict__ sigma,
                                                    float* __restrict__ u) {
  __shared__ Smem smem;

  const int b    = blockIdx.y;
  const int lane = threadIdx.x & 31;
  const int wave = threadIdx.x >> 5;
  const int hi   = lane >> 4;              // K-half selector
  const int lo   = lane & 15;              // row (A) / column (B) index

  const _Float16* hTb = hT + (size_t)b * DF * TT;

#if USE_TDM
  if (wave == 0) tdm_load_tile(hTb, SMEM_HB_OFF);        // prefetch tile 0 -> buf 0
#endif

  for (int i = threadIdx.x; i < TT; i += 128) smem.c[i] = c[(size_t)b * TT + i];

#if USE_TDM
  if (wave == 0) __builtin_amdgcn_s_wait_tensorcnt(0);
#endif
  __syncthreads();

  const float sg     = sigma[0];
  const float inv2s2 = 1.0f / (2.0f * sg * sg);

  const int   s0 = (blockIdx.x * 4 + wave) * 16;
  const float ts = (float)(s0 + lo) + 1.5f;              // grid value of A-row `lo`

  // ---- pass 1: exact row max of score == -(min diff^2) * inv2s2 ----
  float mind2 = 3.4e38f;
  const int tstart = hi << 9;
  for (int t = tstart; t < tstart + 512; ++t) {
    const float diff = ts - smem.c[t];
    mind2 = fminf(mind2, diff * diff);
  }
  mind2 = fminf(mind2, __shfl_xor(mind2, 16, 32));

  // ---- pass 2: fused exp-weights + WMMA + softmax denominator ----
  v8f acc[5];
  const v8f vzero = {0.f, 0.f, 0.f, 0.f, 0.f, 0.f, 0.f, 0.f};
#pragma unroll
  for (int n = 0; n < 5; ++n) acc[n] = vzero;
  float wsum = 0.f;

  auto compute_step = [&](int t0, int buf) {
    // A fragment: lane holds row M=lo; element i -> K = i + 16*hi
    v16h a;
#pragma unroll
    for (int i = 0; i < 16; ++i) {
      const int K = t0 + i + 16 * hi;
      const float diff = ts - smem.c[K];
      const float e = (mind2 - diff * diff) * inv2s2;    // score - rowmax, <= 0
      const float w = __expf(e);
      wsum += w;
      a[i] = (_Float16)w;
    }
#if USE_TDM
    const _Float16* bp = &smem.hb[buf][lo * 32 + 16 * hi];
#pragma unroll
    for (int n = 0; n < 5; ++n) {
      const v16h bf = *(const v16h*)(bp + n * 16 * 32);  // ds_load_b128 x2
      acc[n] = __builtin_amdgcn_wmma_f32_16x16x32_f16(
          false, a, false, bf, (short)0, acc[n], false, false);
    }
#else
    const _Float16* bp = hTb + (size_t)lo * TT + t0 + 16 * hi;
    if (t0 + 32 < TT) __builtin_prefetch(bp + 32, 0, 0);
#pragma unroll
    for (int n = 0; n < 5; ++n) {
      const v16h bf = *(const v16h*)(bp + (size_t)n * 16 * TT);
      acc[n] = __builtin_amdgcn_wmma_f32_16x16x32_f16(
          false, a, false, bf, (short)0, acc[n], false, false);
    }
#endif
    (void)buf;
  };

#if USE_TDM
  for (int t0 = 0; t0 < TT; t0 += 64) {
    // sub-step A: DMA tile t0+32 -> buf1 while computing on buf0
    if (wave == 0 && t0 + 32 < TT)
      tdm_load_tile(hTb + t0 + 32, SMEM_HB_OFF + SMEM_HB_SZ);
    compute_step(t0, 0);
    if (wave == 0) __builtin_amdgcn_s_wait_tensorcnt(0);
    __syncthreads();
    // sub-step B: DMA tile t0+64 -> buf0 while computing on buf1
    if (wave == 0 && t0 + 64 < TT)
      tdm_load_tile(hTb + t0 + 64, SMEM_HB_OFF);
    compute_step(t0 + 32, 1);
    if (wave == 0) __builtin_amdgcn_s_wait_tensorcnt(0);
    __syncthreads();
  }
#else
  for (int t0 = 0; t0 < TT; t0 += 32) compute_step(t0, 0);
#endif

  // combine softmax denominator across the two K-half lanes of each row
  wsum += __shfl_xor(wsum, 16, 32);
  const float rd = 1.0f / wsum;            // lane L: 1/denom of row (L&15)

  // ---- writeout: acc[n][v] is element (row = v + 8*hi, col = n*16 + lo) ----
#pragma unroll
  for (int v = 0; v < 8; ++v) {
    const float rdv  = __shfl(rd, v + 8 * hi, 32);
    const int   srow = s0 + v + 8 * hi;
    float* op = u + ((size_t)b * SZ + srow) * DF + lo;
#pragma unroll
    for (int n = 0; n < 5; ++n) op[n * 16] = acc[n][v] * rdv;
  }
}

// ---------------------------------------------------------------------------
extern "C" void kernel_launch(void* const* d_in, const int* in_sizes, int n_in,
                              void* d_out, int out_size, void* d_ws, size_t ws_size,
                              hipStream_t stream) {
  const float* h     = (const float*)d_in[0];   // [16,1024,80] f32
  const float* d     = (const float*)d_in[1];   // [16,1024]    f32
  const float* sigma = (const float*)d_in[2];   // [1]          f32
  (void)in_sizes; (void)n_in; (void)out_size; (void)ws_size;

  float*    c  = (float*)d_ws;                              // 64 KB
  _Float16* hT = (_Float16*)((char*)d_ws + 64 * 1024);      // 2.62 MB

  centers_kernel<<<dim3(BS), dim3(32), 0, stream>>>(d, c);

  const int tot = BS * DF * TT;
  hT_kernel<<<dim3((tot + 255) / 256), dim3(256), 0, stream>>>(h, hT);

  dim3 grid(SZ / 64, BS);                  // 4 waves/block * 16 s-rows = 64 rows
  align_kernel<<<grid, dim3(128), 0, stream>>>(hT, c, sigma, (float*)d_out);
}
